// SlidingWindowSelfAttention_91250875171632
// MI455X (gfx1250) — compile-verified, matrix operand order checked by reading, not run on hardware
//
#include <hip/hip_runtime.h>

typedef __attribute__((ext_vector_type(16))) __bf16 v16bf;
typedef __attribute__((ext_vector_type(8)))  float  v8f;

#define S_LEN   4096
#define D_MODEL 256
#define N_HEADS 8
#define K_DIM   32
#define WIN     64
#define N_TOK   8192
#define N_TILE  512
#define S_TILE  256

static __device__ __forceinline__ __bf16 f2b(float f) { return (__bf16)f; }

static __device__ __forceinline__ v8f wmma_bf16(v16bf a, v16bf b, v8f c) {
  // D = A(16x32 bf16) * B(32x16 bf16) + C(16x16 f32)
  return __builtin_amdgcn_wmma_f32_16x16x32_bf16(false, a, false, b, (short)0, c, false, false);
}

// Load one 16-lane-half operand stripe: slots 0..7 -> p[0..7], slots 8..15 -> p[16..23]
static __device__ __forceinline__ v16bf load_op(const __bf16* __restrict__ p) {
  v16bf v;
#pragma unroll
  for (int s = 0; s < 8; ++s) { v[s] = p[s]; v[s + 8] = p[s + 16]; }
  return v;
}

static __device__ __forceinline__ v16bf load_op_f32(const float* __restrict__ p) {
  v16bf v;
#pragma unroll
  for (int s = 0; s < 8; ++s) { v[s] = f2b(p[s]); v[s + 8] = f2b(p[s + 16]); }
  return v;
}

// ---------------------------------------------------------------------------
// 1) Weight prep: transpose 256x256 matrices, convert to bf16.
//    WqT/WkT/WvT: [HK][D]   (W is [D][HK]);  WoT: [D][HK]  (Wo is [HK][D])
// ---------------------------------------------------------------------------
__global__ __launch_bounds__(256) void prep_kernel(
    const float* __restrict__ Wq, const float* __restrict__ Wk,
    const float* __restrict__ Wv, const float* __restrict__ Wo,
    __bf16* __restrict__ WqT, __bf16* __restrict__ WkT,
    __bf16* __restrict__ WvT, __bf16* __restrict__ WoT)
{
  const int idx = blockIdx.x * 256 + threadIdx.x;   // 65536 elements
  const int r = idx >> 8;
  const int c = idx & 255;
  WqT[c * 256 + r] = f2b(Wq[idx]);
  WkT[c * 256 + r] = f2b(Wk[idx]);
  WvT[c * 256 + r] = f2b(Wv[idx]);
  WoT[c * 256 + r] = f2b(Wo[idx]);
}

// ---------------------------------------------------------------------------
// 2) QKV projection: [8192 x 256] x [256 x 256] per matrix.
//    One wave per 16-token tile per matrix; 128 WMMAs/wave. All 16 B tiles of
//    a k-chunk are loaded into simultaneously-live registers before the 16
//    consuming WMMAs, so the matrix pipe overlaps the weight fetches instead
//    of stalling on s_wait_loadcnt 0 per WMMA.
// ---------------------------------------------------------------------------
__global__ __launch_bounds__(32) void proj_kernel(
    const float* __restrict__ x,
    const __bf16* __restrict__ WqT, const __bf16* __restrict__ WkT,
    const __bf16* __restrict__ WvT,
    const float* __restrict__ bq, const float* __restrict__ bk,
    const float* __restrict__ bv,
    __bf16* __restrict__ q_ws, __bf16* __restrict__ k_ws,
    __bf16* __restrict__ vT_ws)
{
  const int tile = blockIdx.x;          // 0..511
  const int mat  = blockIdx.y;          // 0:q 1:k 2:v
  const int lane = threadIdx.x;
  const int m    = lane & 15;
  const int half = lane >> 4;

  const __bf16* WT   = (mat == 0) ? WqT : (mat == 1) ? WkT : WvT;
  const float*  bias = (mat == 0) ? bq  : (mat == 1) ? bk  : bv;

  const v8f vzero = {0.f,0.f,0.f,0.f,0.f,0.f,0.f,0.f};
  v8f acc[16];
#pragma unroll
  for (int i = 0; i < 16; ++i) acc[i] = vzero;

  const int token0 = tile * 16;
  const float*  xrow = x  + (token0 + m) * D_MODEL + half * 8;
  const __bf16* wrow = WT + m * D_MODEL + half * 8;

  v16bf a = load_op_f32(xrow);          // kc = 0
#pragma unroll
  for (int kc = 0; kc < 8; ++kc) {
    v16bf bt[16];                       // 16 simultaneously-live B tiles
#pragma unroll
    for (int nt = 0; nt < 16; ++nt) bt[nt] = load_op(wrow + nt * 16 * D_MODEL + kc * 32);
    v16bf anext = a;
    if (kc < 7) anext = load_op_f32(xrow + (kc + 1) * 32);
#pragma unroll
    for (int nt = 0; nt < 16; ++nt) acc[nt] = wmma_bf16(a, bt[nt], acc[nt]);
    a = anext;
  }

#pragma unroll
  for (int nt = 0; nt < 16; ++nt) {
    const int col = m + nt * 16;          // hk index 0..255
    const int h   = col >> 5;
    const int kk  = col & 31;
    const float bb = bias[col];
#pragma unroll
    for (int r = 0; r < 8; ++r) {
      const int token = token0 + r + half * 8;
      const int b  = token >> 12;         // / 4096
      const int sI = token & 4095;
      const float val = acc[nt][r] + bb;
      if (mat == 0)
        q_ws[((b * N_HEADS + h) * S_LEN + sI) * K_DIM + kk] = f2b(val);
      else if (mat == 1)
        k_ws[((b * N_HEADS + h) * S_LEN + sI) * K_DIM + kk] = f2b(val);
      else
        vT_ws[((b * N_HEADS + h) * K_DIM + kk) * S_LEN + sI] = f2b(val);
    }
  }
}

// ---------------------------------------------------------------------------
// 3) Banded attention: one wave per (b,h,query-tile).
//    QK^T WMMAs (pipelined K-tile loads) -> LDS fp32 strip -> softmax with
//    all 32 lanes (2 lanes/row) -> bf16 P -> PV WMMAs.
// ---------------------------------------------------------------------------
__global__ __launch_bounds__(32) void attn_kernel(
    const __bf16* __restrict__ q_ws, const __bf16* __restrict__ k_ws,
    const __bf16* __restrict__ vT_ws, __bf16* __restrict__ o_ws)
{
  __shared__ float  sc[16][160];
  __shared__ __bf16 pb[16][160];

  const int it   = blockIdx.x;          // 0..255 query tile
  const int bh   = blockIdx.y;          // 0..15
  const int b    = bh >> 3;
  const int h    = bh & 7;
  const int lane = threadIdx.x;
  const int m    = lane & 15;
  const int half = lane >> 4;

  const int i0  = it * 16;
  const int jt0 = (it - 4 > 0) ? (it - 4) : 0;
  const int jt1 = (it + 4 < S_TILE - 1) ? (it + 4) : (S_TILE - 1);
  const int ntiles = jt1 - jt0 + 1;     // <= 9
  const int ncols  = ntiles * 16;

  const __bf16* qpl = q_ws  + (b * N_HEADS + h) * S_LEN * K_DIM;
  const __bf16* kpl = k_ws  + (b * N_HEADS + h) * S_LEN * K_DIM;
  const __bf16* vpl = vT_ws + (b * N_HEADS + h) * K_DIM * S_LEN;

  const v8f vzero = {0.f,0.f,0.f,0.f,0.f,0.f,0.f,0.f};

  const v16bf qa = load_op(qpl + (i0 + m) * K_DIM + half * 8);

  v16bf kb = load_op(kpl + (jt0 * 16 + m) * K_DIM + half * 8);
  for (int t = 0; t < ntiles; ++t) {
    v16bf kn = kb;
    if (t + 1 < ntiles)
      kn = load_op(kpl + ((jt0 + t + 1) * 16 + m) * K_DIM + half * 8);
    const v8f sr = wmma_bf16(qa, kb, vzero);
    const int j = (jt0 + t) * 16 + m;   // key column (lane)
#pragma unroll
    for (int r = 0; r < 8; ++r) {
      const int i = i0 + r + half * 8;  // query row
      float v = sr[r] * 0.17677669529663689f;   // 1/sqrt(32)
      const int d = i - j;
      if (d > WIN || d < -WIN) v = -1e9f;
      sc[r + half * 8][t * 16 + m] = v;
    }
    kb = kn;
  }
  __syncthreads();

  { // zero-pad P (columns beyond ncols contribute nothing to PV)
    __bf16* pf = &pb[0][0];
    for (int c = lane; c < 16 * 160; c += 32) pf[c] = f2b(0.f);
  }
  __syncthreads();

  { // exact softmax: 2 lanes per row (interleaved columns), full EXEC
    float mx = -3.4e38f;
    for (int c = half; c < ncols; c += 2) mx = fmaxf(mx, sc[m][c]);
    mx = fmaxf(mx, __shfl_xor(mx, 16, 32));
    float sum = 0.f;
    for (int c = half; c < ncols; c += 2) {
      const float e = __expf(sc[m][c] - mx);
      sc[m][c] = e;
      sum += e;
    }
    sum += __shfl_xor(sum, 16, 32);
    const float inv = 1.f / sum;
    for (int c = half; c < ncols; c += 2) pb[m][c] = f2b(sc[m][c] * inv);
  }
  __syncthreads();

  v8f o0 = vzero, o1 = vzero;
  const int nchunks = (ntiles + 1) >> 1;          // 32-key chunks
  for (int c = 0; c < nchunks; ++c) {
    v16bf pa;
    const int base = c * 32 + half * 8;
#pragma unroll
    for (int s = 0; s < 8; ++s) { pa[s] = pb[m][base + s]; pa[s + 8] = pb[m][base + 16 + s]; }
    v16bf vb0, vb1;
#pragma unroll
    for (int s = 0; s < 16; ++s) {
      const int krel = c * 32 + half * 8 + (s & 7) + ((s >> 3) * 16);
      int j = jt0 * 16 + krel;
      if (j > S_LEN - 1) j = S_LEN - 1;           // padded cols have P==0
      vb0[s] = vpl[m * S_LEN + j];
      vb1[s] = vpl[(m + 16) * S_LEN + j];
    }
    o0 = wmma_bf16(pa, vb0, o0);
    o1 = wmma_bf16(pa, vb1, o1);
  }

  // o_ws: [B*S][256] bf16, col = h*32 + dim
#pragma unroll
  for (int r = 0; r < 8; ++r) {
    const int token = b * S_LEN + i0 + r + half * 8;
    o_ws[token * 256 + h * 32 + m]      = f2b(o0[r]);
    o_ws[token * 256 + h * 32 + m + 16] = f2b(o1[r]);
  }
}

// ---------------------------------------------------------------------------
// 4) Output projection + bias + residual + LayerNorm. One wave per 16 tokens,
//    with the same fully-live B-tile preload as proj_kernel.
// ---------------------------------------------------------------------------
__global__ __launch_bounds__(32) void out_kernel(
    const __bf16* __restrict__ o_ws, const __bf16* __restrict__ WoT,
    const float* __restrict__ bo, const float* __restrict__ x,
    const float* __restrict__ gamma, const float* __restrict__ beta,
    float* __restrict__ out)
{
  const int tile = blockIdx.x;
  const int lane = threadIdx.x;
  const int m    = lane & 15;
  const int half = lane >> 4;
  const int token0 = tile * 16;

  const v8f vzero = {0.f,0.f,0.f,0.f,0.f,0.f,0.f,0.f};
  v8f acc[16];
#pragma unroll
  for (int i = 0; i < 16; ++i) acc[i] = vzero;

  const __bf16* arow = o_ws + (token0 + m) * 256 + half * 8;
  const __bf16* wrow = WoT + m * 256 + half * 8;

  v16bf a = load_op(arow);              // kc = 0
#pragma unroll
  for (int kc = 0; kc < 8; ++kc) {
    v16bf bt[16];                       // 16 simultaneously-live B tiles
#pragma unroll
    for (int nt = 0; nt < 16; ++nt) bt[nt] = load_op(wrow + nt * 16 * 256 + kc * 32);
    v16bf anext = a;
    if (kc < 7) anext = load_op(arow + (kc + 1) * 32);
#pragma unroll
    for (int nt = 0; nt < 16; ++nt) acc[nt] = wmma_bf16(a, bt[nt], acc[nt]);
    a = anext;
  }

#pragma unroll
  for (int r = 0; r < 8; ++r) {
    const int token = token0 + r + half * 8;
    float ssum = 0.f;
#pragma unroll
    for (int nt = 0; nt < 16; ++nt) {
      const int col = m + nt * 16;
      const float y = acc[nt][r] + bo[col] + x[token * 256 + col];
      acc[nt][r] = y;
      ssum += y;
    }
    ssum += __shfl_xor(ssum, 1, 32);
    ssum += __shfl_xor(ssum, 2, 32);
    ssum += __shfl_xor(ssum, 4, 32);
    ssum += __shfl_xor(ssum, 8, 32);
    const float mu = ssum * (1.f / 256.f);
    float vsum = 0.f;
#pragma unroll
    for (int nt = 0; nt < 16; ++nt) { const float d = acc[nt][r] - mu; vsum += d * d; }
    vsum += __shfl_xor(vsum, 1, 32);
    vsum += __shfl_xor(vsum, 2, 32);
    vsum += __shfl_xor(vsum, 4, 32);
    vsum += __shfl_xor(vsum, 8, 32);
    const float rstd = rsqrtf(vsum * (1.f / 256.f) + 1e-3f);
#pragma unroll
    for (int nt = 0; nt < 16; ++nt) {
      const int col = m + nt * 16;
      out[token * 256 + col] = (acc[nt][r] - mu) * rstd * gamma[col] + beta[col];
    }
  }
}

// ---------------------------------------------------------------------------
// Workspace layout (bytes):
//   0        WqT (128 KB) | 131072 WkT | 262144 WvT | 393216 WoT
//   524288   q_ws  (4 MB) | +4MB k_ws | +8MB vT_ws | +12MB o_ws
//   total 17,301,504 bytes
// ---------------------------------------------------------------------------
extern "C" void kernel_launch(void* const* d_in, const int* in_sizes, int n_in,
                              void* d_out, int out_size, void* d_ws, size_t ws_size,
                              hipStream_t stream) {
  (void)in_sizes; (void)n_in; (void)out_size; (void)ws_size;
  const float* x     = (const float*)d_in[0];
  const float* Wq    = (const float*)d_in[1];
  const float* bq    = (const float*)d_in[2];
  const float* Wk    = (const float*)d_in[3];
  const float* bk    = (const float*)d_in[4];
  const float* Wv    = (const float*)d_in[5];
  const float* bv    = (const float*)d_in[6];
  const float* Wo    = (const float*)d_in[7];
  const float* bo    = (const float*)d_in[8];
  const float* gamma = (const float*)d_in[9];
  const float* beta  = (const float*)d_in[10];
  float* out = (float*)d_out;

  char* ws = (char*)d_ws;
  __bf16* WqT  = (__bf16*)(ws + 0);
  __bf16* WkT  = (__bf16*)(ws + 131072);
  __bf16* WvT  = (__bf16*)(ws + 262144);
  __bf16* WoT  = (__bf16*)(ws + 393216);
  __bf16* q_ws = (__bf16*)(ws + 524288);
  __bf16* k_ws = (__bf16*)(ws + 524288 + 1ull * 4194304);
  __bf16* vT_ws= (__bf16*)(ws + 524288 + 2ull * 4194304);
  __bf16* o_ws = (__bf16*)(ws + 524288 + 3ull * 4194304);

  prep_kernel<<<256, 256, 0, stream>>>(Wq, Wk, Wv, Wo, WqT, WkT, WvT, WoT);
  proj_kernel<<<dim3(N_TILE, 3), 32, 0, stream>>>(x, WqT, WkT, WvT, bq, bk, bv,
                                                  q_ws, k_ws, vT_ws);
  attn_kernel<<<dim3(S_TILE, 2 * N_HEADS), 32, 0, stream>>>(q_ws, k_ws, vT_ws, o_ws);
  out_kernel<<<N_TILE, 32, 0, stream>>>(o_ws, WoT, bo, x, gamma, beta, out);
}
